// TemplateEmbedding_32169305047622
// MI455X (gfx1250) — matrix operand order repaired and checked by reading, not compiled, not run on hardware
//
#include <hip/hip_runtime.h>
#include <hip/hip_bf16.h>

// ---------------------------------------------------------------------------
// AlphaFold template embedding on gfx1250 (MI455X).
// All dense GEMMs (projections, triangle einsums, transition, output head) go
// through a WMMA (v_wmma_f32_16x16x32_f16) tile kernel with branchless,
// vectorized loads and a template-specialized (branch-free) epilogue;
// elementwise / softmax / frame math on VALU. Working set ~55MB -> L2 resident.
// ---------------------------------------------------------------------------

#define R      128   // N_RES
#define CP     64    // C_PAIR
#define QCH    128   // Q_CH
#define NH     4     // N_HEAD
#define DH     16    // D_HEAD
#define NTPL   4     // N_TEMPL
#define NATOM  24
#define GLY    7
#define CA_IDX 1
#define CB_IDX 4

typedef __attribute__((ext_vector_type(16))) _Float16 v16h;
typedef __attribute__((ext_vector_type(8)))  float    v8f;

__device__ __forceinline__ float sigf(float x) { return 1.0f / (1.0f + expf(-x)); }

// ---------------------------------------------------------------------------
// Batched, strided GEMM: D[m,n] (+)= sum_k A[m,k]*B[k,n], fp32 in/out, f16
// WMMA inner product. Requires M % 64 == 0, N % 16 == 0, K in {64,128}
// (guaranteed by all call sites -> zero bounds checks, unconditional loads,
// branch-free body).
// Block = (32,4): 4 waves handle 4 consecutive M tiles; grid = (M/64, N/16, batch).
// AK1: A is K-contiguous (sAk==1) -> per-lane A slots form two 8-float runs,
// loaded as four float4 (global_load_b128).
// ACC: D += result (8 unconditional RMW loads batched under one wait).
// VGPR layouts per CDNA5 ISA 7.12.2:
//  A (16x32 f16): lanes0-15 rows M0-15 hold K {0..7,16..23}; lanes16-31 hold
//                 K {8..15,24..31}.  slot e -> K = e + (e>=8?8:0) + (lane>=16?8:0)
//  B (32x16 f16): N = lane&15; slot e -> K = e + (e>=8?8:0) + (lane>=16?8:0), same as A
//  C/D (16x16 f32, 8 VGPRs): N = lane&15; VGPR v -> M = v + (lane>=16 ? 8 : 0)
// ---------------------------------------------------------------------------
template <int K, bool AK1, bool ACC>
__global__ __launch_bounds__(128) void k_gemm(
    const float* __restrict__ A, const float* __restrict__ B, float* __restrict__ D,
    int sAm, int sAk, int sBk, int sBn, int sDm, int sDn,
    int bA, int bB, int bD)
{
    const int lane = threadIdx.x;                       // 0..31 (wave32)
    const int tm = (blockIdx.x * 4 + threadIdx.y) * 16;
    const int tn = blockIdx.y * 16;

    const int kAo = (lane >> 4) << 3;                   // 0 or 8
    const int kBo = (lane >> 4) << 4;                   // 0 or 16

    const float* Ap = A + (long long)blockIdx.z * bA + (long long)(tm + (lane & 15)) * sAm;
    const float* Bp = B + (long long)blockIdx.z * bB + (long long)(tn + (lane & 15)) * sBn;

    v8f c = {};
#pragma unroll
    for (int k0 = 0; k0 < K; k0 += 32) {
        v16h af, bf;
        if constexpr (AK1) {
            const float4* p = reinterpret_cast<const float4*>(Ap + k0 + kAo);
            const float4 r0 = p[0];   // K offsets 0..3
            const float4 r1 = p[1];   // 4..7
            const float4 r2 = p[4];   // 16..19
            const float4 r3 = p[5];   // 20..23
            af[0]=(_Float16)r0.x;  af[1]=(_Float16)r0.y;  af[2]=(_Float16)r0.z;  af[3]=(_Float16)r0.w;
            af[4]=(_Float16)r1.x;  af[5]=(_Float16)r1.y;  af[6]=(_Float16)r1.z;  af[7]=(_Float16)r1.w;
            af[8]=(_Float16)r2.x;  af[9]=(_Float16)r2.y;  af[10]=(_Float16)r2.z; af[11]=(_Float16)r2.w;
            af[12]=(_Float16)r3.x; af[13]=(_Float16)r3.y; af[14]=(_Float16)r3.z; af[15]=(_Float16)r3.w;
        } else {
#pragma unroll
            for (int e = 0; e < 16; ++e)
                af[e] = (_Float16)Ap[(k0 + kAo + (e < 8 ? e : e + 8)) * sAk];
        }
#pragma unroll
        for (int e = 0; e < 16; ++e)
            bf[e] = (_Float16)Bp[(k0 + kAo + (e < 8 ? e : e + 8)) * sBk];

        c = __builtin_amdgcn_wmma_f32_16x16x32_f16(false, af, false, bf,
                                                   (short)0, c, false, false);
    }

    float* Dp = D + (long long)blockIdx.z * bD
                  + (long long)tm * sDm + (long long)(tn + (lane & 15)) * sDn;
    const int mo = (lane >> 4) << 3;
    if constexpr (ACC) {
        float old[8];
#pragma unroll
        for (int v = 0; v < 8; ++v) old[v] = Dp[(long long)(mo + v) * sDm];
#pragma unroll
        for (int v = 0; v < 8; ++v) Dp[(long long)(mo + v) * sDm] = old[v] + c[v];
    } else {
#pragma unroll
        for (int v = 0; v < 8; ++v) Dp[(long long)(mo + v) * sDm] = c[v];
    }
}

// tiny N=4 bias projection (x @ Wb), negligible FLOPs -> plain VALU
__global__ void k_projN4(const float* __restrict__ X, const float* __restrict__ W,
                         float* __restrict__ out)
{
    const int i = blockIdx.x * blockDim.x + threadIdx.x;   // R*R*NH threads
    if (i >= R * R * NH) return;
    const int row = i >> 2, n = i & 3;
    float acc = 0.0f;
#pragma unroll 8
    for (int k = 0; k < CP; ++k) acc += X[row * CP + k] * W[k * NH + n];
    out[i] = acc;
}

// ---------------------------------------------------------------------------
// LayerNorm over last dim C (rows = gridDim.x, blockDim.x == C, C in {64,128})
// ---------------------------------------------------------------------------
__global__ void k_layernorm(const float* __restrict__ x, const float* __restrict__ sc,
                            const float* __restrict__ bi, float* __restrict__ y, int C)
{
    __shared__ float red[128];
    const long long row = blockIdx.x;
    const int c = threadIdx.x;
    const float v = x[row * C + c];
    red[c] = v; __syncthreads();
    for (int st = C >> 1; st > 0; st >>= 1) { if (c < st) red[c] += red[c + st]; __syncthreads(); }
    const float mu = red[0] / (float)C; __syncthreads();
    const float d = v - mu;
    red[c] = d * d; __syncthreads();
    for (int st = C >> 1; st > 0; st >>= 1) { if (c < st) red[c] += red[c + st]; __syncthreads(); }
    const float var = red[0] / (float)C;
    y[row * C + c] = d * rsqrtf(var + 1e-5f) * sc[c] + bi[c];
}

// elementwise helpers -------------------------------------------------------
__global__ void k_gatemul(const float* __restrict__ g, const float* __restrict__ p,
                          const float* __restrict__ mask, float* __restrict__ out,
                          long long total)
{
    const long long i = (long long)blockIdx.x * blockDim.x + threadIdx.x;
    if (i < total) out[i] = sigf(g[i]) * p[i] * mask[i / CP];
}

__global__ void k_sigmul(const float* __restrict__ g, float* __restrict__ o, long long total)
{
    const long long i = (long long)blockIdx.x * blockDim.x + threadIdx.x;
    if (i < total) o[i] = sigf(g[i]) * o[i];
}

__global__ void k_sigmul_add(float* __restrict__ act, const float* __restrict__ g,
                             const float* __restrict__ t, long long total)
{
    const long long i = (long long)blockIdx.x * blockDim.x + threadIdx.x;
    if (i < total) act[i] += sigf(g[i]) * t[i];
}

__global__ void k_silu_mul(const float* __restrict__ h, float* __restrict__ out, long long total)
{
    const long long i = (long long)blockIdx.x * blockDim.x + threadIdx.x;
    if (i < total) {
        const long long row = i >> 7;       // /128
        const int c = (int)(i & 127);
        const float a = h[row * 256 + c];
        const float b = h[row * 256 + 128 + c];
        out[i] = a * sigf(a) * b;
    }
}

__global__ void k_add(float* __restrict__ dst, const float* __restrict__ src, long long total)
{
    const long long i = (long long)blockIdx.x * blockDim.x + threadIdx.x;
    if (i < total) dst[i] += src[i];
}

__global__ void k_transpose(const float* __restrict__ src, float* __restrict__ dst)
{
    const long long i = (long long)blockIdx.x * blockDim.x + threadIdx.x;
    const long long total = (long long)R * R * CP;
    if (i < total) {
        const int c = (int)(i % CP);
        const long long rij = i / CP;
        const int jj = (int)(rij % R);
        const int ii = (int)(rij / R);
        dst[((long long)jj * R + ii) * CP + c] = src[i];
    }
}

__global__ void k_add_transpose(float* __restrict__ act, const float* __restrict__ delta)
{
    const long long i = (long long)blockIdx.x * blockDim.x + threadIdx.x;
    const long long total = (long long)R * R * CP;
    if (i < total) {
        const int c = (int)(i % CP);
        const long long rij = i / CP;
        const int jj = (int)(rij % R);
        const int ii = (int)(rij / R);
        act[i] += delta[((long long)jj * R + ii) * CP + c];
    }
}

__global__ void k_relu_scale(const float* __restrict__ src, float* __restrict__ dst,
                             float scale, long long total)
{
    const long long i = (long long)blockIdx.x * blockDim.x + threadIdx.x;
    if (i < total) dst[i] = fmaxf(src[i] * scale, 0.0f);
}

// ---------------------------------------------------------------------------
// Per-residue template prep: pseudo-beta, masks, backbone frame (rot), CA pos.
// Record stride 20 floats: [0:3]=pb, [3]=pb_mask, [4]=rmask, [5:8]=p1,
// [8:17]=rot (row-major rot[m][k], columns = e0,e1,e2)
// ---------------------------------------------------------------------------
__global__ void k_prep(const int* __restrict__ aat, const float* __restrict__ pos,
                       const float* __restrict__ amask, const int* __restrict__ table,
                       float* __restrict__ res)
{
    const int i = threadIdx.x;
    if (i >= R) return;
    const int a = aat[i];
    const int* bb = table + a * 8 * 3;    // group 0 row
    const int ci = bb[0], bi = bb[1], ai = bb[2];

    float p0[3], p1[3], p2[3], pb[3];
    float mA, mB, mC, pbm;
    {
        mA = amask[i * NATOM + ai];
        mB = amask[i * NATOM + bi];
        mC = amask[i * NATOM + ci];
        for (int k = 0; k < 3; ++k) {
            p0[k] = pos[(i * NATOM + ai) * 3 + k] * mA;
            p1[k] = pos[(i * NATOM + bi) * 3 + k] * mB;
            p2[k] = pos[(i * NATOM + ci) * 3 + k] * mC;
        }
        const int pidx = (a == GLY) ? CA_IDX : CB_IDX;
        pbm = amask[i * NATOM + pidx];
        for (int k = 0; k < 3; ++k) pb[k] = pos[(i * NATOM + pidx) * 3 + k] * pbm;
    }
    const float rmask = mA * mB * mC;

    float v1[3], v2[3];
    for (int k = 0; k < 3; ++k) { v1[k] = p2[k] - p1[k]; v2[k] = p0[k] - p1[k]; }
    const float n1 = rsqrtf(v1[0]*v1[0] + v1[1]*v1[1] + v1[2]*v1[2] + 1e-6f);
    float e0[3] = { v1[0]*n1, v1[1]*n1, v1[2]*n1 };
    const float cd = e0[0]*v2[0] + e0[1]*v2[1] + e0[2]*v2[2];
    float w[3] = { v2[0]-cd*e0[0], v2[1]-cd*e0[1], v2[2]-cd*e0[2] };
    const float n2 = rsqrtf(w[0]*w[0] + w[1]*w[1] + w[2]*w[2] + 1e-6f);
    float e1[3] = { w[0]*n2, w[1]*n2, w[2]*n2 };
    float e2[3] = { e0[1]*e1[2]-e0[2]*e1[1], e0[2]*e1[0]-e0[0]*e1[2], e0[0]*e1[1]-e0[1]*e1[0] };

    float* r = res + i * 20;
    r[0]=pb[0]; r[1]=pb[1]; r[2]=pb[2]; r[3]=pbm; r[4]=rmask;
    r[5]=p1[0]; r[6]=p1[1]; r[7]=p1[2];
    for (int m = 0; m < 3; ++m) { r[8+m*3+0]=e0[m]; r[8+m*3+1]=e1[m]; r[8+m*3+2]=e2[m]; }
}

// ---------------------------------------------------------------------------
// act init: distogram one-hot @ E0, pb2d @ E1, aatype one-hots @ E2/E3,
// unit-vector channels @ E4..E6, bb2d @ E7, plus q_term.
// grid = R*R blocks, block = 64 threads (channel c).
// ---------------------------------------------------------------------------
__global__ void k_init_act(const float* __restrict__ res, const int* __restrict__ aat,
                           const float* __restrict__ qterm, const float* __restrict__ mc2d,
                           const float* __restrict__ E0, const float* __restrict__ E1,
                           const float* __restrict__ E2, const float* __restrict__ E3,
                           const float* __restrict__ E4, const float* __restrict__ E5,
                           const float* __restrict__ E6, const float* __restrict__ E7,
                           float* __restrict__ act)
{
    const int ij = blockIdx.x;
    const int i = ij / R, j = ij % R;
    const int c = threadIdx.x;
    const float* ri = res + i * 20;
    const float* rj = res + j * 20;

    const float dx = ri[0]-rj[0], dy = ri[1]-rj[1], dz = ri[2]-rj[2];
    const float d2 = dx*dx + dy*dy + dz*dz;
    const float mc = mc2d[ij];
    const float pb2 = ri[3] * rj[3] * mc;
    const float bb2 = ri[4] * rj[4] * mc;

    int bin = -1;
    for (int k = 0; k < 39; ++k) {
        float lo = 3.25f + 1.25f * k; lo *= lo;
        float hi;
        if (k == 38) hi = 1e8f;
        else { hi = 3.25f + 1.25f * (k + 1); hi *= hi; }
        if (d2 > lo && d2 < hi) bin = k;
    }

    const float dm0 = rj[5]-ri[5], dm1 = rj[6]-ri[6], dm2 = rj[7]-ri[7];
    float u0 = ri[8+0]*dm0 + ri[8+3]*dm1 + ri[8+6]*dm2;   // sum_m rot[m][0]*d[m]
    float u1 = ri[8+1]*dm0 + ri[8+4]*dm1 + ri[8+7]*dm2;
    float u2 = ri[8+2]*dm0 + ri[8+5]*dm1 + ri[8+8]*dm2;
    const float inv = rsqrtf(u0*u0 + u1*u1 + u2*u2 + 1e-6f) * bb2;
    u0 *= inv; u1 *= inv; u2 *= inv;

    const int aI = aat[i], aJ = aat[j];
    const long long idx = (long long)ij * CP + c;
    float v = qterm[idx];
    if (bin >= 0) v += pb2 * E0[bin * CP + c];
    v += pb2 * E1[c];
    v += E2[aJ * CP + c] + E3[aI * CP + c];
    v += u0 * E4[c] + u1 * E5[c] + u2 * E6[c];
    v += bb2 * E7[c];
    act[idx] = v;
}

// ---------------------------------------------------------------------------
// Fused triangle attention core: one block per (i,j,h), 128 threads = key idx.
// q/k/v/xb layouts: q[(i*R+j)*NH*DH + h*DH + d], xb[(j*R+t)*NH + h].
// mtrans selects mask[i,t] vs mask[t,i] (for "ending" variant on act^T).
// ---------------------------------------------------------------------------
__global__ void k_attn(const float* __restrict__ q, const float* __restrict__ kk,
                       const float* __restrict__ vv, const float* __restrict__ xb,
                       const float* __restrict__ mask, int mtrans, float* __restrict__ o)
{
    __shared__ float qs[DH];
    __shared__ float red[R];
    __shared__ float ps[R];
    const int i = blockIdx.x / R;
    const int j = blockIdx.x % R;
    const int h = blockIdx.y;
    const int t = threadIdx.x;

    if (t < DH) qs[t] = q[(((long long)i * R + j) * NH + h) * DH + t] * 0.25f; // D_HEAD^-0.5
    __syncthreads();

    const float* kp = kk + (((long long)i * R + t) * NH + h) * DH;
    float lg = 0.0f;
#pragma unroll
    for (int d = 0; d < DH; ++d) lg += qs[d] * kp[d];
    lg += xb[((long long)j * R + t) * NH + h];
    const float mv = mtrans ? mask[t * R + i] : mask[i * R + t];
    lg += 1e9f * (mv - 1.0f);

    red[t] = lg; __syncthreads();
    for (int st = R >> 1; st > 0; st >>= 1) { if (t < st) red[t] = fmaxf(red[t], red[t + st]); __syncthreads(); }
    const float mx = red[0]; __syncthreads();
    const float e = expf(lg - mx);
    red[t] = e; __syncthreads();
    for (int st = R >> 1; st > 0; st >>= 1) { if (t < st) red[t] += red[t + st]; __syncthreads(); }
    const float s = red[0]; __syncthreads();
    ps[t] = e / s; __syncthreads();

    if (t < DH) {
        float acc = 0.0f;
        for (int k2 = 0; k2 < R; ++k2)
            acc += ps[k2] * vv[(((long long)i * R + k2) * NH + h) * DH + t];
        o[(((long long)i * R + j) * NH + h) * DH + t] = acc;
    }
}

// ===========================================================================
// Host orchestration
// ===========================================================================
struct TriP   { const float *Wag,*Wap,*Wbg,*Wbp,*Wg,*Wo,*ln_b,*ln_s,*oln_b,*oln_s; };
struct AttnP  { const float *Wb,*Wg,*Wk,*Wo,*Wq,*Wv,*ln_b,*ln_s; };
struct TransP { const float *W1,*W2,*ln_b,*ln_s; };
struct BlockP { AttnP attn_end, attn_start; TransP trans; TriP tri_in, tri_out; };

struct Ctx {
    hipStream_t s;
    float *qterm,*sum,*act,*x,*t0,*t1,*t2,*t3,*actt,*big,*xb,*res;
    const float *pad2d,*mc2d;
};

static const long long SZ = (long long)R * R * CP;   // 1,048,576 floats

// M % 64 == 0, N % 16 == 0, K in {64,128} required.
static inline void gemm(hipStream_t s, const float* A, const float* B, float* D,
                        int M, int N, int K,
                        int sAm, int sAk, int sBk, int sBn, int sDm, int sDn,
                        int batch, int bA, int bB, int bD, int accum)
{
    dim3 g(M / 64, N / 16, batch);
    dim3 b(32, 4);
    if (K == 64) {
        if (sAk == 1) {
            if (accum) k_gemm<64, true , true ><<<g, b, 0, s>>>(A, B, D, sAm, sAk, sBk, sBn, sDm, sDn, bA, bB, bD);
            else       k_gemm<64, true , false><<<g, b, 0, s>>>(A, B, D, sAm, sAk, sBk, sBn, sDm, sDn, bA, bB, bD);
        } else {
            if (accum) k_gemm<64, false, true ><<<g, b, 0, s>>>(A, B, D, sAm, sAk, sBk, sBn, sDm, sDn, bA, bB, bD);
            else       k_gemm<64, false, false><<<g, b, 0, s>>>(A, B, D, sAm, sAk, sBk, sBn, sDm, sDn, bA, bB, bD);
        }
    } else {
        if (sAk == 1) {
            if (accum) k_gemm<128, true , true ><<<g, b, 0, s>>>(A, B, D, sAm, sAk, sBk, sBn, sDm, sDn, bA, bB, bD);
            else       k_gemm<128, true , false><<<g, b, 0, s>>>(A, B, D, sAm, sAk, sBk, sBn, sDm, sDn, bA, bB, bD);
        } else {
            if (accum) k_gemm<128, false, true ><<<g, b, 0, s>>>(A, B, D, sAm, sAk, sBk, sBn, sDm, sDn, bA, bB, bD);
            else       k_gemm<128, false, false><<<g, b, 0, s>>>(A, B, D, sAm, sAk, sBk, sBn, sDm, sDn, bA, bB, bD);
        }
    }
}

// row-major projection: D[M,N] (+)= X[M,K] @ W[K,N]
static inline void proj(hipStream_t s, const float* X, const float* W, float* D,
                        int M, int N, int K, int accum)
{
    gemm(s, X, W, D, M, N, K, K, 1, N, 1, N, 1, 1, 0, 0, 0, accum);
}

static inline dim3 eg(long long total) { return dim3((unsigned)((total + 255) / 256)); }

static void run_tri(Ctx& c, const TriP& p, bool outgoing)
{
    const int RR = R * R;
    k_layernorm<<<RR, CP, 0, c.s>>>(c.act, p.ln_s, p.ln_b, c.x, CP);
    proj(c.s, c.x, p.Wag, c.t2, RR, CP, CP, 0);
    proj(c.s, c.x, p.Wap, c.t3, RR, CP, CP, 0);
    k_gatemul<<<eg(SZ), 256, 0, c.s>>>(c.t2, c.t3, c.pad2d, c.t0, SZ);
    proj(c.s, c.x, p.Wbg, c.t2, RR, CP, CP, 0);
    proj(c.s, c.x, p.Wbp, c.t3, RR, CP, CP, 0);
    k_gatemul<<<eg(SZ), 256, 0, c.s>>>(c.t2, c.t3, c.pad2d, c.t1, SZ);
    proj(c.s, c.x, p.Wg, c.t2, RR, CP, CP, 0);     // gate, kept in t2

    const int RC = R * CP;                          // 8192
    if (outgoing)   // o[i,j,c] = sum_k a[i,k,c]*b[j,k,c]
        gemm(c.s, c.t0, c.t1, c.t3, R, R, 128, RC, CP, CP, RC, RC, CP, CP, 1, 1, 1, 0);
    else            // o[i,j,c] = sum_k a[k,i,c]*b[k,j,c]
        gemm(c.s, c.t0, c.t1, c.t3, R, R, 128, CP, RC, RC, CP, RC, CP, CP, 1, 1, 1, 0);

    k_layernorm<<<RR, CP, 0, c.s>>>(c.t3, p.oln_s, p.oln_b, c.t3, CP);
    proj(c.s, c.t3, p.Wo, c.t0, RR, CP, CP, 0);
    k_sigmul_add<<<eg(SZ), 256, 0, c.s>>>(c.act, c.t2, c.t0, SZ);
}

static void run_attn(Ctx& c, const AttnP& p, bool starting)
{
    const int RR = R * R;
    const float* src = c.act;
    if (!starting) { k_transpose<<<eg(SZ), 256, 0, c.s>>>(c.act, c.actt); src = c.actt; }
    k_layernorm<<<RR, CP, 0, c.s>>>(src, p.ln_s, p.ln_b, c.x, CP);
    proj(c.s, c.x, p.Wq, c.t0, RR, CP, CP, 0);
    proj(c.s, c.x, p.Wk, c.t1, RR, CP, CP, 0);
    proj(c.s, c.x, p.Wv, c.t2, RR, CP, CP, 0);
    proj(c.s, c.x, p.Wg, c.t3, RR, CP, CP, 0);
    k_projN4<<<eg((long long)RR * NH), 256, 0, c.s>>>(c.x, p.Wb, c.xb);

    dim3 ag(RR, NH);
    k_attn<<<ag, R, 0, c.s>>>(c.t0, c.t1, c.t2, c.xb, c.pad2d, starting ? 0 : 1, c.big);
    k_sigmul<<<eg(SZ), 256, 0, c.s>>>(c.t3, c.big, SZ);   // o *= sigmoid(gate)

    if (starting) {
        proj(c.s, c.big, p.Wo, c.act, RR, CP, CP, 1);
    } else {
        proj(c.s, c.big, p.Wo, c.actt, RR, CP, CP, 0);
        k_add_transpose<<<eg(SZ), 256, 0, c.s>>>(c.act, c.actt);
    }
}

static void run_trans(Ctx& c, const TransP& p)
{
    const int RR = R * R;
    k_layernorm<<<RR, CP, 0, c.s>>>(c.act, p.ln_s, p.ln_b, c.x, CP);
    proj(c.s, c.x, p.W1, c.big, RR, 4 * CP, CP, 0);                 // 16384 x 256
    const long long half = (long long)RR * 128;
    k_silu_mul<<<eg(half), 256, 0, c.s>>>(c.big, c.t0, half);       // 16384 x 128
    proj(c.s, c.t0, p.W2, c.act, RR, CP, 128, 1);
}

extern "C" void kernel_launch(void* const* d_in, const int* in_sizes, int n_in,
                              void* d_out, int out_size, void* d_ws, size_t ws_size,
                              hipStream_t stream)
{
    (void)in_sizes; (void)n_in; (void)out_size; (void)ws_size;
    const int RR = R * R;

    // --- raw inputs (setup_inputs() order) ---------------------------------
    const float* query = (const float*)d_in[0];      // (128,128,128)
    const int*   aat   = (const int*)  d_in[1];      // (4,128)
    const float* pos   = (const float*)d_in[2];      // (4,128,24,3)
    const float* amask = (const float*)d_in[3];      // (4,128,24)
    const float* pad2d = (const float*)d_in[4];      // (128,128)
    const float* mc2d  = (const float*)d_in[5];      // (128,128)
    const int*   table = (const int*)  d_in[6];      // (31,8,3)

    // --- params: JAX pytree leaves, dict keys in sorted order --------------
    // top level: Wout, blocks, emb, oln_b, oln_s, qln_b, qln_s
    // per block: attn_end, attn_start, trans, tri_in, tri_out
    int pi = 7;
    auto F = [&]() { return (const float*)d_in[pi++]; };

    const float* Wout = F();
    BlockP blk[2];
    for (int b = 0; b < 2; ++b) {
        AttnP* aps[2] = { &blk[b].attn_end, &blk[b].attn_start };
        for (int a = 0; a < 2; ++a) {
            AttnP& ap = *aps[a];
            ap.Wb = F(); ap.Wg = F(); ap.Wk = F(); ap.Wo = F();
            ap.Wq = F(); ap.Wv = F(); ap.ln_b = F(); ap.ln_s = F();
        }
        blk[b].trans.W1 = F(); blk[b].trans.W2 = F();
        blk[b].trans.ln_b = F(); blk[b].trans.ln_s = F();
        TriP* tps[2] = { &blk[b].tri_in, &blk[b].tri_out };
        for (int t = 0; t < 2; ++t) {
            TriP& tp = *tps[t];
            tp.Wag = F(); tp.Wap = F(); tp.Wbg = F(); tp.Wbp = F();
            tp.Wg = F(); tp.Wo = F(); tp.ln_b = F(); tp.ln_s = F();
            tp.oln_b = F(); tp.oln_s = F();
        }
    }
    const float* E[9];
    for (int e = 0; e < 9; ++e) E[e] = F();
    const float* oln_b = F(); const float* oln_s = F();
    const float* qln_b = F(); const float* qln_s = F();

    // --- workspace layout (floats); total ~13.07M * 4B + small ≈ 55 MB -----
    float* w = (float*)d_ws;
    Ctx c;
    c.s     = stream;
    c.qterm = w + 0 * SZ;
    c.sum   = w + 1 * SZ;
    c.act   = w + 2 * SZ;
    c.x     = w + 3 * SZ;
    c.t0    = w + 4 * SZ;
    c.t1    = w + 5 * SZ;
    c.t2    = w + 6 * SZ;
    c.t3    = w + 7 * SZ;
    c.actt  = w + 8 * SZ;
    c.big   = w + 9 * SZ;                    // 4*SZ floats (16 MB)
    c.xb    = w + 13 * SZ;                   // RR*NH floats
    c.res   = c.xb + (long long)RR * NH;     // 128*20 floats
    c.pad2d = pad2d; c.mc2d = mc2d;

    // --- prologue: q_term = LN(query) @ E8 ---------------------------------
    (void)hipMemsetAsync(c.sum, 0, (size_t)SZ * sizeof(float), stream);
    k_layernorm<<<RR, QCH, 0, stream>>>(query, qln_s, qln_b, c.big, QCH);   // big as xq_ln
    proj(stream, c.big, E[8], c.qterm, RR, CP, QCH, 0);

    // --- template loop ------------------------------------------------------
    for (int t = 0; t < NTPL; ++t) {
        k_prep<<<1, 128, 0, stream>>>(aat + t * R, pos + (long long)t * R * NATOM * 3,
                                      amask + (long long)t * R * NATOM, table, c.res);
        k_init_act<<<RR, CP, 0, stream>>>(c.res, aat + t * R, c.qterm, mc2d,
                                          E[0], E[1], E[2], E[3], E[4], E[5], E[6], E[7],
                                          c.act);
        for (int b = 0; b < 2; ++b) {
            run_tri (c, blk[b].tri_out,    true);
            run_tri (c, blk[b].tri_in,     false);
            run_attn(c, blk[b].attn_start, true);
            run_attn(c, blk[b].attn_end,   false);
            run_trans(c, blk[b].trans);
        }
        k_layernorm<<<RR, CP, 0, stream>>>(c.act, oln_s, oln_b, c.x, CP);
        k_add<<<eg(SZ), 256, 0, stream>>>(c.sum, c.x, SZ);
    }

    // --- epilogue: relu(sum / (1e-7 + 4)) @ Wout -> d_out (16384 x 128) ----
    k_relu_scale<<<eg(SZ), 256, 0, stream>>>(c.sum, c.x, 1.0f / (1e-7f + 4.0f), SZ);
    proj(stream, c.x, Wout, (float*)d_out, RR, QCH, CP, 0);
}
